// NeighborCooccurrenceEncoder_11227044512145
// MI455X (gfx1250) — matrix operand order repaired and checked by reading, not compiled, not run on hardware
//
#include <hip/hip_runtime.h>

typedef __attribute__((ext_vector_type(16))) _Float16 v16h;
typedef __attribute__((ext_vector_type(8)))  _Float16 v8h;
typedef __attribute__((ext_vector_type(2)))  _Float16 v2h;
typedef __attribute__((ext_vector_type(8)))  float    v8f;

namespace {
constexpr int kS        = 512;   // neighbors per row
constexpr int kD        = 64;    // feature dim
constexpr int kB        = 256;   // batch
constexpr int kVocabPad = 128;   // VOCAB=100 padded
constexpr int kW2Stride = 72;    // padded halves per W2^T row (bank-conflict pad)
}

__global__ __launch_bounds__(256)
void ncoe_fused(const int* __restrict__ src_ids,
                const int* __restrict__ dst_ids,
                const float* __restrict__ W1,   // [2,64] row-major (in,out)
                const float* __restrict__ b1,   // [64]
                const float* __restrict__ W2,   // [64,64] row-major (in,out)
                const float* __restrict__ b2,   // [64]
                float* __restrict__ out)        // [src BxSx64 | dst BxSx64]
{
  __shared__ int sh_sid[kS];
  __shared__ int sh_did[kS];
  __shared__ int sh_hs[kVocabPad];
  __shared__ int sh_hd[kVocabPad];
  __shared__ __align__(16) _Float16 sh_w2[kD * kW2Stride];  // W2^T as f16: [n][k]
  __shared__ v2h   sh_wa2[kD / 2];  // (W1[0][2p], W1[0][2p+1]) as f16
  __shared__ v2h   sh_wb2[kD / 2];  // (W1[1][2p], W1[1][2p+1]) as f16
  __shared__ v2h   sh_b12[kD / 2];  // (b1[2p],   b1[2p+1])   as f16
  __shared__ float sh_b2[kD];

  const int tid  = threadIdx.x;
  const int blk  = blockIdx.x;
  const int b    = blk >> 1;      // batch row
  const int part = blk & 1;       // 0: src queries, 1: dst queries
  const int lane = tid & 31;
  const int wave = tid >> 5;
  const int hi   = (lane >> 4) & 1;
  const int ln   = lane & 15;

  // ---------------- Phase 1: stage ids + weights into LDS ----------------
  const int* sp = src_ids + (size_t)b * kS;
  const int* dp = dst_ids + (size_t)b * kS;
  const int sid0 = sp[tid], sid1 = sp[tid + 256];
  const int did0 = dp[tid], did1 = dp[tid + 256];
  sh_sid[tid] = sid0;  sh_sid[tid + 256] = sid1;
  sh_did[tid] = did0;  sh_did[tid + 256] = did1;
  if (tid < kVocabPad) { sh_hs[tid] = 0; sh_hd[tid] = 0; }
  for (int idx = tid; idx < kD * kD; idx += 256) {
    const int k = idx >> 6;          // input dim
    const int n = idx & 63;          // output dim
    sh_w2[n * kW2Stride + k] = (_Float16)W2[idx];
  }
  if (tid < kD / 2) {
    const int c = 2 * tid;
    sh_wa2[tid] = (v2h){(_Float16)W1[c],      (_Float16)W1[c + 1]};
    sh_wb2[tid] = (v2h){(_Float16)W1[kD + c], (_Float16)W1[kD + c + 1]};
    sh_b12[tid] = (v2h){(_Float16)b1[c],      (_Float16)b1[c + 1]};
  }
  if (tid < kD) sh_b2[tid] = b2[tid];
  __syncthreads();

  // ---------------- Phase 2: per-row histograms (VOCAB=100) ----------------
  atomicAdd(&sh_hs[sid0], 1);
  atomicAdd(&sh_hs[sid1], 1);
  atomicAdd(&sh_hd[did0], 1);
  atomicAdd(&sh_hd[did1], 1);
  __syncthreads();

  // ------------- Hoist W2 B-fragments (loop-invariant per wave) -------------
  // B layout (32x16 f16): lane l element e -> K = (e<8 ? e : e+8) + 8*(l>=16), N = l%16
  v16h bf[2][4];
  #pragma unroll
  for (int kk = 0; kk < 2; ++kk) {
    #pragma unroll
    for (int nt = 0; nt < 4; ++nt) {
      const v8h* p = (const v8h*)&sh_w2[(nt * 16 + ln) * kW2Stride + kk * 32 + hi * 8];
      bf[kk][nt] = __builtin_shufflevector(p[0], p[2],
          0, 1, 2, 3, 4, 5, 6, 7, 8, 9, 10, 11, 12, 13, 14, 15);
    }
  }

  // ---- Hoist per-lane W1/b1 coefficient pairs into registers (invariant) ----
  // A-fragment dword j of chunk kk covers columns c, c+1 with
  //   c = 2j + (j>=4 ? 8 : 0) + 8*hi + 32*kk   -> pair index p = c/2
  v2h rwa[16], rwb[16], rb1[16];
  #pragma unroll
  for (int kk = 0; kk < 2; ++kk) {
    #pragma unroll
    for (int j = 0; j < 8; ++j) {
      const int p = j + (j >= 4 ? 4 : 0) + hi * 4 + kk * 16;
      rwa[kk * 8 + j] = sh_wa2[p];
      rwb[kk * 8 + j] = sh_wb2[p];
      rb1[kk * 8 + j] = sh_b12[p];
    }
  }

  // ---- b2 folded into the WMMA C operand: per-lane splat (col = 16*nt + ln) ----
  v8f cinit[4];
  #pragma unroll
  for (int nt = 0; nt < 4; ++nt) {
    const float bb = sh_b2[nt * 16 + ln];
    v8f c;
    #pragma unroll
    for (int r = 0; r < 8; ++r) c[r] = bb;
    cinit[nt] = c;
  }

  // -------- Prefetch the 4 tiles' (cs, cd) counts (batch the LDS reads) --------
  float csA[4], cdA[4];
  #pragma unroll
  for (int i = 0; i < 4; ++i) {
    const int t  = part * 32 + wave * 4 + i;
    const int qr = t * 16 + ln;                // this lane's A-fragment row
    const int id = (qr < kS) ? sh_sid[qr] : sh_did[qr - kS];
    const float msk = (id != 0) ? 1.f : 0.f;   // padding-id mask on features
    csA[i] = (float)sh_hs[id] * msk;
    cdA[i] = (float)sh_hd[id] * msk;
  }

  // ---------------- Phase 3: fused MLP via WMMA, 4 tiles per wave ----------------
  #pragma unroll
  for (int i = 0; i < 4; ++i) {
    const _Float16 csh = (_Float16)csA[i];
    const _Float16 cdh = (_Float16)cdA[i];
    const v2h cs2 = (v2h){csh, csh};
    const v2h cd2 = (v2h){cdh, cdh};

    v8f acc[4];
    #pragma unroll
    for (int kk = 0; kk < 2; ++kk) {
      union { v16h v; v2h p[8]; } a;
      #pragma unroll
      for (int j = 0; j < 8; ++j) {
        v2h hp = cs2 * rwa[kk * 8 + j] + (cd2 * rwb[kk * 8 + j] + rb1[kk * 8 + j]);
        a.p[j] = __builtin_elementwise_max(hp, (v2h)0);   // ReLU
      }
      #pragma unroll
      for (int nt = 0; nt < 4; ++nt) {
        const v8f cin = (kk == 0) ? cinit[nt] : acc[nt];  // bias preloaded as C
        acc[nt] = __builtin_amdgcn_wmma_f32_16x16x32_f16(
            false, a.v, false, bf[kk][nt], (short)0, cin, false, false);
      }
    }

    // D layout: lane l VGPR r -> row r + 8*(l>=16), col 16*nt + l%16
    const int t    = part * 32 + wave * 4 + i;
    const int qb   = t * 16;
    const int side = (qb >= kS) ? 1 : 0;
    const size_t base = (size_t)side * ((size_t)kB * kS * kD)
                      + ((size_t)b * kS + (size_t)(qb - side * kS)) * kD
                      + (size_t)(hi * 8) * kD + (size_t)ln;
    float* op = out + base;
    #pragma unroll
    for (int nt = 0; nt < 4; ++nt) {
      #pragma unroll
      for (int r = 0; r < 8; ++r)
        op[r * kD + nt * 16] = acc[nt][r];
    }
  }
}

extern "C" void kernel_launch(void* const* d_in, const int* in_sizes, int n_in,
                              void* d_out, int out_size, void* d_ws, size_t ws_size,
                              hipStream_t stream) {
  (void)in_sizes; (void)n_in; (void)out_size; (void)d_ws; (void)ws_size;
  const int*   src = (const int*)d_in[0];
  const int*   dst = (const int*)d_in[1];
  const float* W1  = (const float*)d_in[2];
  const float* b1  = (const float*)d_in[3];
  const float* W2  = (const float*)d_in[4];
  const float* b2  = (const float*)d_in[5];
  float* outp = (float*)d_out;
  dim3 grid(kB * 2), block(256);
  hipLaunchKernelGGL(ncoe_fused, grid, block, 0, stream,
                     src, dst, W1, b1, W2, b2, outp);
}
